// SAWindowCALayer_63522566308360
// MI455X (gfx1250) — compile-verified, matrix operand order checked by reading, not run on hardware
//
#include <hip/hip_runtime.h>
#include <hip/hip_bf16.h>
#include <math.h>

typedef __bf16 bf16;
typedef __attribute__((ext_vector_type(16))) __bf16 v16bf;
typedef __attribute__((ext_vector_type(8)))  float  v8f;

#define B_    4
#define S_    1024
#define DM    512
#define DBAR  256
#define G3    768
#define TOCT  1024
#define CON   32
#define H_    8
#define DH    32

// ---------------------------------------------------------------------------
// WMMA helpers (CDNA5: v_wmma_f32_16x16x32_bf16, wave32)
// A (M=16 x K=32) row-major, lane = tid&31: row = lane&15,
//   VGPR0-3 = K[kb..kb+7], VGPR4-7 = K[kb+16..kb+23], kb = 8*(lane>=16).
// B operand = W (N x K) row-major, same lane pattern (lane holds column N).
// ---------------------------------------------------------------------------
__device__ __forceinline__ v16bf load_frag(const bf16* __restrict__ base,
                                           int row0, int ld, int k0) {
  int lane = threadIdx.x & 31;
  const bf16* p = base + (size_t)(row0 + (lane & 15)) * ld + k0 + ((lane >> 4) << 3);
  v16bf f;
#pragma unroll
  for (int i = 0; i < 8; ++i) f[i] = p[i];
#pragma unroll
  for (int i = 0; i < 8; ++i) f[i + 8] = p[i + 16];
  return f;
}

__device__ __forceinline__ v8f wmma_bf16(v16bf a, v16bf b, v8f c) {
  return __builtin_amdgcn_wmma_f32_16x16x32_bf16(false, a, false, b, (short)0, c,
                                                 false, false);
}

__device__ __forceinline__ void store_tile(v8f acc, const float* __restrict__ bias,
                                           float* __restrict__ Cf,
                                           bf16* __restrict__ Cb, int ldc,
                                           int row0, int col0) {
  int lane = threadIdx.x & 31;
  int col = col0 + (lane & 15);
  float bv = bias ? bias[col] : 0.f;
  int rbase = row0 + ((lane >> 4) << 3);
#pragma unroll
  for (int i = 0; i < 8; ++i) {
    float v = acc[i] + bv;
    size_t off = (size_t)(rbase + i) * ldc + col;
    if (Cf) Cf[off] = v;
    if (Cb) Cb[off] = (bf16)v;
  }
}

__device__ __forceinline__ float sigmoidf_(float x) {
  return 1.0f / (1.0f + __expf(-x));
}

// ---------------------------------------------------------------------------
// f32 -> bf16 conversion (grid-stride)
// ---------------------------------------------------------------------------
__global__ void f32_to_bf16_kernel(const float* __restrict__ src,
                                   bf16* __restrict__ dst, int n) {
  int i = blockIdx.x * blockDim.x + threadIdx.x;
  int stride = gridDim.x * blockDim.x;
  for (; i < n; i += stride) dst[i] = (bf16)src[i];
}

// ---------------------------------------------------------------------------
// avg-pool(8) over onset + sinusoidal PE -> bf16  [B, 1024, 32]
// ---------------------------------------------------------------------------
__global__ void pool_pe_kernel(const float* __restrict__ onset,
                               bf16* __restrict__ pp) {
  int idx = blockIdx.x * blockDim.x + threadIdx.x;
  if (idx >= B_ * TOCT * CON) return;
  int c = idx & 31;
  int t = (idx >> 5) & (TOCT - 1);
  int b = idx >> 15;
  const float* src = onset + ((size_t)b * 8192 + (size_t)t * 8) * CON + c;
  float s = 0.f;
#pragma unroll
  for (int p = 0; p < 8; ++p) s += src[p * CON];
  s *= 0.125f;
  int i = c >> 1;
  float div = __expf(-(float)i * (9.210340371976184f / 16.0f)); // ln(1e4)/half
  float ang = (float)t * div;
  float pe = (c & 1) ? __cosf(ang) : __sinf(ang);
  pp[idx] = (bf16)(s + pe);
}

// ---------------------------------------------------------------------------
// WMMA GEMM: C[M,N] = A[M,K] @ W[N,K]^T + bias.
// One wave computes a 16x64 block (4 N-tiles) -> A fragment reused 4x.
// ---------------------------------------------------------------------------
__global__ __launch_bounds__(256) void gemm_bf16_kernel(
    const bf16* __restrict__ A, int lda, const bf16* __restrict__ W, int ldb,
    const float* __restrict__ bias, float* __restrict__ Cf,
    bf16* __restrict__ Cb, int ldc, int Mtiles, int NGroups, int Kslices) {
  int wid = (blockIdx.x * blockDim.x + threadIdx.x) >> 5;
  int nw = (gridDim.x * blockDim.x) >> 5;
  int total = Mtiles * NGroups;
  for (int t = wid; t < total; t += nw) {
    int mt = t / NGroups, ng = t - mt * NGroups;
    int row0 = mt << 4, col0 = ng << 6;  // group of 4 N-tiles
    v8f acc0 = {}, acc1 = {}, acc2 = {}, acc3 = {};
    for (int ks = 0; ks < Kslices; ++ks) {
      if (ks + 1 < Kslices)
        __builtin_prefetch(W + (size_t)col0 * ldb + ((ks + 1) << 5), 0, 1);
      v16bf a = load_frag(A, row0, lda, ks << 5);
      v16bf b0 = load_frag(W, col0, ldb, ks << 5);
      v16bf b1 = load_frag(W, col0 + 16, ldb, ks << 5);
      v16bf b2 = load_frag(W, col0 + 32, ldb, ks << 5);
      v16bf b3 = load_frag(W, col0 + 48, ldb, ks << 5);
      acc0 = wmma_bf16(a, b0, acc0);
      acc1 = wmma_bf16(a, b1, acc1);
      acc2 = wmma_bf16(a, b2, acc2);
      acc3 = wmma_bf16(a, b3, acc3);
    }
    store_tile(acc0, bias, Cf, Cb, ldc, row0, col0);
    store_tile(acc1, bias, Cf, Cb, ldc, row0, col0 + 16);
    store_tile(acc2, bias, Cf, Cb, ldc, row0, col0 + 32);
    store_tile(acc3, bias, Cf, Cb, ldc, row0, col0 + 48);
  }
}

// ---------------------------------------------------------------------------
// Sequential dual-GRU scan. One block, 256 threads (8 waves), all 4 batches.
// Per matmul each wave registers all 8 A K-slice fragments once, then streams
// weight fragments from L2: inner loop is pure global-load + WMMA.
// ---------------------------------------------------------------------------
__global__ __launch_bounds__(256) void gru_scan_kernel(
    const float* __restrict__ gi_note,   // [B*S*768], includes bih_note
    const bf16* __restrict__ Whh_note,   // [768x256]
    const float* __restrict__ bhh_note,  // [768]
    const bf16* __restrict__ Wih_bar,    // [768x256]
    const float* __restrict__ bih_bar,   // [768]
    const bf16* __restrict__ Whh_bar,    // [768x256]
    const float* __restrict__ bhh_bar,   // [768]
    const unsigned char* __restrict__ mask,  // [B*S] bool
    float* __restrict__ out_carried,     // [B*S*256]
    float* __restrict__ out_final,       // [B*256]
    bf16* __restrict__ g_bf_out) {       // [B*S*256]
  __shared__ bf16 hA[16 * DBAR];   // A tile for h (rows 4..15 stay zero)
  __shared__ bf16 gA[16 * DBAR];   // A tile for g
  __shared__ float hF[B_ * DBAR];
  __shared__ float gF[B_ * DBAR];
  __shared__ float gate[3][B_ * G3];  // gh_note, gi_bar, gh_bar

  int tid = threadIdx.x;
  int lane = tid & 31;
  int wv = tid >> 5;

  for (int i = tid; i < 16 * DBAR; i += 256) { hA[i] = (bf16)0.f; gA[i] = (bf16)0.f; }
  for (int i = tid; i < B_ * DBAR; i += 256) { hF[i] = 0.f; gF[i] = 0.f; }
  __syncthreads();

  for (int s = 0; s < S_; ++s) {
    // --- recurrent matmuls [4x256]@[256x768] x3 (M padded to 16) ---
    for (int m = 0; m < 3; ++m) {
      const bf16* Asrc = (m == 2) ? gA : hA;
      const bf16* Bw = (m == 0) ? Whh_note : (m == 1) ? Wih_bar : Whh_bar;
      v16bf af[8];
#pragma unroll
      for (int ks = 0; ks < 8; ++ks) af[ks] = load_frag(Asrc, 0, DBAR, ks << 5);
      for (int ct = wv; ct < 48; ct += 8) {
        int col0 = ct << 4;
        if (ct + 8 < 48)
          __builtin_prefetch(Bw + (size_t)(col0 + 128) * DBAR, 0, 1);
        v8f acc = {};
#pragma unroll
        for (int ks = 0; ks < 8; ++ks) {
          v16bf b = load_frag(Bw, col0, DBAR, ks << 5);
          acc = wmma_bf16(af[ks], b, acc);
        }
        if (lane < 16) {  // rows 0..3 live in VGPR 0..3 of lanes 0..15
          int col = col0 + lane;
#pragma unroll
          for (int i = 0; i < 4; ++i) gate[m][i * G3 + col] = acc[i];
        }
      }
    }
    __syncthreads();

    // --- gate math + masked latch: thread j handles channel j for 4 batches ---
    int j = tid;
#pragma unroll
    for (int b = 0; b < B_; ++b) {
      const float* gi = gi_note + ((size_t)(b * S_ + s)) * G3;
      float hprev = hF[b * DBAR + j];
      float gprev = gF[b * DBAR + j];
      // note GRU (input side precomputed)
      float r1 = sigmoidf_(gi[j] + gate[0][b * G3 + j] + bhh_note[j]);
      float z1 = sigmoidf_(gi[DBAR + j] + gate[0][b * G3 + DBAR + j] + bhh_note[DBAR + j]);
      float n1 = tanhf(gi[2 * DBAR + j] +
                       r1 * (gate[0][b * G3 + 2 * DBAR + j] + bhh_note[2 * DBAR + j]));
      float hnew = (1.f - z1) * n1 + z1 * hprev;
      // bar GRU (x = previous h_note)
      float gir = gate[1][b * G3 + j] + bih_bar[j];
      float giz = gate[1][b * G3 + DBAR + j] + bih_bar[DBAR + j];
      float gin = gate[1][b * G3 + 2 * DBAR + j] + bih_bar[2 * DBAR + j];
      float r2 = sigmoidf_(gir + gate[2][b * G3 + j] + bhh_bar[j]);
      float z2 = sigmoidf_(giz + gate[2][b * G3 + DBAR + j] + bhh_bar[DBAR + j]);
      float n2 = tanhf(gin + r2 * (gate[2][b * G3 + 2 * DBAR + j] + bhh_bar[2 * DBAR + j]));
      float gnew = (1.f - z2) * n2 + z2 * gprev;
      bool mm = mask[b * S_ + s] != 0;
      float gout = mm ? gnew : gprev;
      float hout = mm ? 0.f : hnew;
      hF[b * DBAR + j] = hout;
      gF[b * DBAR + j] = gout;
      hA[b * DBAR + j] = (bf16)hout;
      gA[b * DBAR + j] = (bf16)gout;
      size_t oi = ((size_t)(b * S_ + s)) * DBAR + j;
      out_carried[oi] = gout;
      g_bf_out[oi] = (bf16)gout;
      if (s == S_ - 1) out_final[b * DBAR + j] = gout;
    }
    __syncthreads();
  }
}

// ---------------------------------------------------------------------------
// Attention: block = one (b,h,128-row s-block); 8 waves each own a 16-row s-strip.
// K strip (1024x32 bf16, 64KB) staged through LDS in 16KB double-buffered
// chunks via global_load_async_to_lds_b128 (ASYNCcnt), overlapping DMA of
// chunk c+1 with WMMA + online softmax on chunk c. 8x less L2 traffic.
// ---------------------------------------------------------------------------
__global__ __launch_bounds__(256) void attn_kernel(const bf16* __restrict__ Qb,
                                                   const bf16* __restrict__ Kb,
                                                   const float* __restrict__ pbs_ptr,
                                                   float* __restrict__ com_h) {
  __shared__ bf16 Kbuf[2][256 * 32];     // 2 x 16KB chunks (256 t-rows x 32)
  __shared__ float tile_s[8][16][17];
  int tid = threadIdx.x;
  int lane = tid & 31;
  int w = tid >> 5;
  int blk = blockIdx.x;                  // 256 blocks
  int b = blk >> 6;
  int h = (blk >> 3) & (H_ - 1);
  int sblk = blk & 7;
  int s0 = sblk * 128 + w * 16;
  float pbs = pbs_ptr[0];
  const float inv_sqrt = 0.17677669529663687f;  // 1/sqrt(32)
  const bf16* qbase = Qb + (size_t)b * S_ * DBAR + h * DH;
  const bf16* kbase = Kb + (size_t)b * TOCT * DBAR + h * DH;
  v16bf aq = load_frag(qbase, s0, DBAR, 0);  // K=32 in one fragment

  // async DMA of one 256-row chunk: 1024 x 16B segments across 256 threads
  auto issue_chunk = [&](int c) {
    int buf = c & 1;
#pragma unroll
    for (int i = 0; i < 4; ++i) {
      int seg = i * 256 + tid;
      int row = seg >> 2;
      int part = seg & 3;
      unsigned long long gp =
          (unsigned long long)(kbase + (size_t)(c * 256 + row) * DBAR + part * 8);
      unsigned ldsa = (unsigned)(size_t)&Kbuf[buf][row * 32 + part * 8];
      asm volatile("global_load_async_to_lds_b128 %0, %1, off"
                   :: "v"(ldsa), "v"(gp) : "memory");
    }
  };

  float m = -3.0e38f, s1 = 0.f, stp = 0.f;
  issue_chunk(0);
  for (int c = 0; c < 4; ++c) {
    asm volatile("s_wait_asynccnt 0x0" ::: "memory");
    __syncthreads();                      // all waves' DMA visible, prev reads done
    if (c + 1 < 4) issue_chunk(c + 1);
    const bf16* kl = Kbuf[c & 1];
    for (int j2 = 0; j2 < 16; ++j2) {
      v16bf bfr = load_frag(kl, j2 << 4, 32, 0);
      v8f acc = {};
      acc = wmma_bf16(aq, bfr, acc);
      int r = (lane >> 4) << 3;
#pragma unroll
      for (int i = 0; i < 8; ++i) tile_s[w][r + i][lane & 15] = acc[i];
      // same-wave LDS ops are in-order: reads below see the writes above
      if (lane < 16) {
        float vals[16];
        float tmax = -3.0e38f;
        int t0 = c * 256 + (j2 << 4);
#pragma unroll
        for (int cc = 0; cc < 16; ++cc) {
          float tp = (float)(t0 + cc) * (1.0f / 1023.0f);
          float v = tile_s[w][lane][cc] * inv_sqrt - pbs * tp;
          vals[cc] = v;
          tmax = fmaxf(tmax, v);
        }
        float nm = fmaxf(m, tmax);
        float sc = __expf(m - nm);
        s1 *= sc; stp *= sc;
#pragma unroll
        for (int cc = 0; cc < 16; ++cc) {
          float e = __expf(vals[cc] - nm);
          s1 += e;
          stp += e * (float)(t0 + cc) * (1.0f / 1023.0f);
        }
        m = nm;
      }
    }
  }
  if (lane < 16)
    com_h[((size_t)(b * H_ + h)) * S_ + s0 + lane] = stp / s1;
}

// ---------------------------------------------------------------------------
// com = mean over heads, apply bar_mask
// ---------------------------------------------------------------------------
__global__ void finalize_kernel(const float* __restrict__ com_h,
                                const unsigned char* __restrict__ mask,
                                float* __restrict__ out) {
  int idx = blockIdx.x * blockDim.x + threadIdx.x;
  if (idx >= B_ * S_) return;
  int b = idx >> 10, s = idx & (S_ - 1);
  float acc = 0.f;
#pragma unroll
  for (int h = 0; h < H_; ++h) acc += com_h[((size_t)(b * H_ + h)) * S_ + s];
  acc *= (1.0f / (float)H_);
  out[idx] = mask[idx] ? acc : 0.f;
}

// ---------------------------------------------------------------------------
extern "C" void kernel_launch(void* const* d_in, const int* in_sizes, int n_in,
                              void* d_out, int out_size, void* d_ws, size_t ws_size,
                              hipStream_t stream) {
  (void)in_sizes; (void)n_in; (void)out_size; (void)ws_size;
  const float* token    = (const float*)d_in[0];
  const float* onset    = (const float*)d_in[1];
  const unsigned char* mask = (const unsigned char*)d_in[2];
  const float* Wih_note = (const float*)d_in[3];
  const float* Whh_note = (const float*)d_in[4];
  const float* bih_note = (const float*)d_in[5];
  const float* bhh_note = (const float*)d_in[6];
  const float* Wih_bar  = (const float*)d_in[7];
  const float* Whh_bar  = (const float*)d_in[8];
  const float* bih_bar  = (const float*)d_in[9];
  const float* bhh_bar  = (const float*)d_in[10];
  const float* Wq       = (const float*)d_in[11];
  const float* bq       = (const float*)d_in[12];
  const float* Wk       = (const float*)d_in[13];
  const float* bk       = (const float*)d_in[14];
  const float* pbs      = (const float*)d_in[15];

  float* out = (float*)d_out;
  float* out_com     = out;            // [B*S]     = 4096
  float* out_final   = out + 4096;     // [1,B,256] = 1024
  float* out_carried = out + 5120;     // [B,S,256] = 1048576

  char* ws = (char*)d_ws;
  size_t off = 0;
  auto take = [&](size_t bytes) -> void* {
    void* p = ws + off;
    off += (bytes + 255) & ~(size_t)255;
    return p;
  };
  bf16* Xb   = (bf16*)take((size_t)B_ * S_ * DM * 2);
  bf16* Wnb  = (bf16*)take((size_t)G3 * DM * 2);
  bf16* Whnb = (bf16*)take((size_t)G3 * DBAR * 2);
  bf16* Wibb = (bf16*)take((size_t)G3 * DBAR * 2);
  bf16* Whbb = (bf16*)take((size_t)G3 * DBAR * 2);
  bf16* Wqb  = (bf16*)take((size_t)DBAR * DBAR * 2);
  bf16* Wkb  = (bf16*)take((size_t)DBAR * CON * 2);
  bf16* PPb  = (bf16*)take((size_t)B_ * TOCT * CON * 2);
  bf16* Kbf  = (bf16*)take((size_t)B_ * TOCT * DBAR * 2);
  bf16* Qbf  = (bf16*)take((size_t)B_ * S_ * DBAR * 2);
  bf16* Gbf  = (bf16*)take((size_t)B_ * S_ * DBAR * 2);
  float* gi  = (float*)take((size_t)B_ * S_ * G3 * 4);
  float* comh = (float*)take((size_t)B_ * H_ * S_ * 4);

  // 1) convert to bf16
  f32_to_bf16_kernel<<<1024, 256, 0, stream>>>(token, Xb, B_ * S_ * DM);
  f32_to_bf16_kernel<<<384, 256, 0, stream>>>(Wih_note, Wnb, G3 * DM);
  f32_to_bf16_kernel<<<192, 256, 0, stream>>>(Whh_note, Whnb, G3 * DBAR);
  f32_to_bf16_kernel<<<192, 256, 0, stream>>>(Wih_bar, Wibb, G3 * DBAR);
  f32_to_bf16_kernel<<<192, 256, 0, stream>>>(Whh_bar, Whbb, G3 * DBAR);
  f32_to_bf16_kernel<<<64, 256, 0, stream>>>(Wq, Wqb, DBAR * DBAR);
  f32_to_bf16_kernel<<<8, 256, 0, stream>>>(Wk, Wkb, DBAR * CON);

  // 2) pooled onset + PE
  pool_pe_kernel<<<512, 256, 0, stream>>>(onset, PPb);

  // 3) gi_note = X @ Wih_note^T + bih_note   (M=4096,N=768,K=512)
  gemm_bf16_kernel<<<256, 256, 0, stream>>>(Xb, DM, Wnb, DM, bih_note, gi,
                                            nullptr, G3, 256, 12, 16);
  // 4) K = (pool+pe) @ Wk^T + bk             (M=4096,N=256,K=32)
  gemm_bf16_kernel<<<256, 256, 0, stream>>>(PPb, CON, Wkb, CON, bk, nullptr,
                                            Kbf, DBAR, 256, 4, 1);
  // 5) sequential dual-GRU scan
  gru_scan_kernel<<<1, 256, 0, stream>>>(gi, Whnb, bhh_note, Wibb, bih_bar,
                                         Whbb, bhh_bar, mask, out_carried,
                                         out_final, Gbf);
  // 6) Q = h_bar_carried @ Wq^T + bq         (M=4096,N=256,K=256)
  gemm_bf16_kernel<<<256, 256, 0, stream>>>(Gbf, DBAR, Wqb, DBAR, bq, nullptr,
                                            Qbf, DBAR, 256, 4, 8);
  // 7) attention (async-LDS staged K) -> per-head com
  attn_kernel<<<256, 256, 0, stream>>>(Qbf, Kbf, pbs, comh);
  // 8) mean over heads + mask
  finalize_kernel<<<16, 256, 0, stream>>>(comh, mask, out_com);
}
